// SKConv2d_75402445848599
// MI455X (gfx1250) — matrix operand order repaired and checked by reading, not compiled, not run on hardware
//
#include <hip/hip_runtime.h>
#include <hip/hip_bf16.h>

typedef __attribute__((ext_vector_type(16))) __bf16 v16bf;
typedef __attribute__((ext_vector_type(8)))  __bf16 v8bf;
typedef __attribute__((ext_vector_type(8)))  float  v8f;

#define B_    32
#define C_    128
#define H_    56
#define W_    56
#define O_    256
#define T_    6
#define R_    8
#define R2_   72
#define D_    1152      // C*3*3
#define HP_   58        // H + 2*pad
#define WP_   58
#define HW_   3136      // H*W
#define NPIX_ 100352    // B*H*W

#define MBLK       64   // output pixels per block
#define KSTEP      32   // WMMA K per step
#define NKB        (D_ / KSTEP)   // 36 K-steps (even)
#define LDS_STRIDE 40   // bf16 elems per LDS row (80B, 16B aligned)

// ---------------------------------------------------------------------------
// Kernel 1: W_eff, K reordered tap-major: k' = tap*C + c  (tap = kh*3+kw)
//   Wt[o][k'] = (1/T) * sum_t( S1[t]@U1[t] + U2[t]^T@S2[t] )[c*9+tap][o]
// ---------------------------------------------------------------------------
__global__ void build_weight(const float* __restrict__ S1s,
                             const float* __restrict__ U1s,
                             const float* __restrict__ U2s,
                             const float* __restrict__ S2s,
                             __bf16* __restrict__ Wt) {
  int idx = blockIdx.x * blockDim.x + threadIdx.x;
  if (idx >= O_ * D_) return;
  int o   = idx / D_;
  int kp  = idx - o * D_;        // tap-major K index
  int tap = kp / C_;
  int c   = kp - tap * C_;
  int d   = c * 9 + tap;         // original (C, kh, kw)-order index
  float acc = 0.f;
  for (int t = 0; t < T_; ++t) {
    const float* s1 = S1s + (t * D_ + d) * R_;   // S1s[t, d, :]
    const float* u1 = U1s + t * R_ * O_ + o;     // U1s[t, :, o]
    float a = 0.f;
#pragma unroll
    for (int r = 0; r < R_; ++r) a += s1[r] * u1[r * O_];
    const float* u2 = U2s + t * R2_ * D_ + d;    // U2s[t, :, d]
    const float* s2 = S2s + t * R2_ * O_ + o;    // S2s[t, :, o]
    float b = 0.f;
    for (int r = 0; r < R2_; ++r) b += u2[r * D_] * s2[r * O_];
    acc += a + b;
  }
  Wt[o * D_ + kp] = (__bf16)(acc * (1.0f / T_));
}

// ---------------------------------------------------------------------------
// Kernel 2: x (f32 NCHW) -> xpad (bf16 padded NHWC [B][58][58][C]).
// One block per (b, hp) row; LDS transpose so reads (over w) and writes
// (over c) are both coalesced. Halo rows/cols written as zeros.
// ---------------------------------------------------------------------------
#define TS_ 58  // LDS row stride (58*2B -> 29-bank stride, conflict-free)
__global__ __launch_bounds__(256)
void pad_nhwc(const float* __restrict__ x, __bf16* __restrict__ xpad) {
  int bhp = blockIdx.x;
  int b   = bhp / HP_;
  int hp  = bhp - b * HP_;
  __bf16* orow = xpad + (size_t)((b * HP_ + hp) * WP_) * C_;  // [WP][C]
  const int rowElems = WP_ * C_;                              // 7424
  if (hp == 0 || hp == HP_ - 1) {
    for (int i = threadIdx.x; i < rowElems; i += 256) orow[i] = (__bf16)0.f;
    return;
  }
  __shared__ __bf16 lds[C_ * TS_];
  int h = hp - 1;
  const float* xrow = x + (size_t)b * C_ * HW_ + h * W_;
  for (int i = threadIdx.x; i < C_ * W_; i += 256) {   // coalesced f32 reads
    int c = i / W_;
    int w = i - c * W_;
    lds[c * TS_ + w] = (__bf16)xrow[c * HW_ + w];
  }
  __syncthreads();
  for (int j = threadIdx.x; j < rowElems; j += 256) {  // coalesced bf16 writes
    int wp = j / C_;
    int c  = j - wp * C_;
    __bf16 v = (__bf16)0.f;
    if (wp >= 1 && wp <= W_) v = lds[c * TS_ + (wp - 1)];
    orow[j] = v;
  }
}

// ---------------------------------------------------------------------------
// GEMM helpers
// ---------------------------------------------------------------------------
__device__ __forceinline__ void load_afrag(v16bf (&af)[8],
                                           const __bf16* __restrict__ wrow,
                                           int kb) {
#pragma unroll
  for (int t = 0; t < 8; ++t) {
    const __bf16* wt = wrow + kb + t * 16 * D_;
    union { v16bf v; v8bf h[2]; } au;
    au.h[0] = *(const v8bf*)(wt);        // global_load_b128
    au.h[1] = *(const v8bf*)(wt + 8);    // global_load_b128
    af[t] = au.v;
  }
}

__device__ __forceinline__ void issue_stage(const __bf16* __restrict__ xpad,
                                            int kb2, unsigned ldst,
                                            unsigned voff) {
  int tap2 = kb2 >> 7;                   // kb/128: one tap per K-block
  int c02  = kb2 & (C_ - 1);
  int kh2  = tap2 / 3;
  int kw2  = tap2 - kh2 * 3;
  const __bf16* sbase = xpad + ((kh2 * WP_ + kw2) * C_ + c02);  // uniform
  asm volatile("global_load_async_to_lds_b128 %0, %1, %2"
               :
               : "v"(ldst), "v"(voff), "s"(sbase)
               : "memory");
}

// ---------------------------------------------------------------------------
// Kernel 3: implicit-im2col GEMM (out^T tiling):
//   D[o, m] = sum_k' Wt[o, k'] * X[k', m]
// X staging: double-buffered global_load_async_to_lds_b128 (ASYNCcnt).
// A fragments: manually double-buffered register arrays, loop unrolled x2 so
// the rotation is copy-free (no v_mov shuffles between load and WMMA regs).
// Block: 256 thr = 8 waves = 4 m-groups(16 pix) x 2 o-groups(128 ch).
// ---------------------------------------------------------------------------
__global__ __launch_bounds__(256)
void skconv_gemm(const __bf16* __restrict__ Wt,
                 const __bf16* __restrict__ xpad,
                 const float*  __restrict__ bias,
                 float* __restrict__ out) {
  __shared__ __align__(16) __bf16 ldsX[2][MBLK * LDS_STRIDE];

  const int tid  = threadIdx.x;
  const int wave = tid >> 5;     // 0..7
  const int lane = tid & 31;
  const int hl   = lane >> 4;    // half-wave: 0 or 1
  const int l16  = lane & 15;

  const int mg     = wave & 3;   // m-group: rows mg*16 .. mg*16+15 of the tile
  const int og     = wave >> 2;  // o-group: 0 -> o 0..127, 1 -> o 128..255
  const int oBase  = og * 128;
  const int mBlock = blockIdx.x * MBLK;

  // ---- per-thread staging addresses (fixed; per-K-step delta is scalar) ----
  const int ml = tid >> 2;       // 0..63: pixel within tile
  const int q  = tid & 3;        // channel octet 0..3 (8 bf16 = 16B)
  int sm  = mBlock + ml;
  int sb  = sm / HW_;
  int shw = sm - sb * HW_;
  int sh  = shw / W_;
  int sw  = shw - sh * W_;
  const unsigned voff =
      (unsigned)((((sb * HP_ + sh) * WP_ + sw) * C_ + q * 8) * 2);
  const unsigned lds0 =
      (unsigned)(uintptr_t)&ldsX[0][ml * LDS_STRIDE + q * 8];  // low32 = LDS off
  const unsigned lds1 =
      (unsigned)(uintptr_t)&ldsX[1][ml * LDS_STRIDE + q * 8];

  v8f acc[8];
#pragma unroll
  for (int t = 0; t < 8; ++t) acc[t] = (v8f){};

  const __bf16* wrow = Wt + (oBase + l16) * D_ + hl * 16;  // + kb per step

  // ---- preamble: stage K-step 0 into buffer 0; preload A frags for step 0
  asm volatile("global_load_async_to_lds_b128 %0, %1, %2"
               :
               : "v"(lds0), "v"(voff), "s"(xpad)
               : "memory");
  v16bf af0[8], af1[8];
  load_afrag(af0, wrow, 0);

  for (int i = 0; i < NKB; i += 2) {
    const int kb = i * KSTEP;

    // ================= even step i: consume af0 / ldsX[0] =================
    load_afrag(af1, wrow, kb + KSTEP);          // next step's W (always valid)
    asm volatile("s_wait_asynccnt 0" ::: "memory");
    __syncthreads();
    issue_stage(xpad, kb + KSTEP, lds1, voff);  // stage step i+1 -> buf 1
    {
      const __bf16* xr = &ldsX[0][(mg * 16 + l16) * LDS_STRIDE + hl * 16];
      v16bf bfrag = *(const v16bf*)xr;          // 2x ds_load_b128
#pragma unroll
      for (int t = 0; t < 8; ++t)
        acc[t] = __builtin_amdgcn_wmma_f32_16x16x32_bf16(
            false, af0[t], false, bfrag, (short)0, acc[t], false, false);
    }

    // ================= odd step i+1: consume af1 / ldsX[1] ================
    const int kbn = (i + 2 < NKB) ? kb + 2 * KSTEP : 0;  // dead on last pass
    load_afrag(af0, wrow, kbn);
    asm volatile("s_wait_asynccnt 0" ::: "memory");
    __syncthreads();
    issue_stage(xpad, kbn, lds0, voff);         // stage step i+2 -> buf 0
    {
      const __bf16* xr = &ldsX[1][(mg * 16 + l16) * LDS_STRIDE + hl * 16];
      v16bf bfrag = *(const v16bf*)xr;          // 2x ds_load_b128
#pragma unroll
      for (int t = 0; t < 8; ++t)
        acc[t] = __builtin_amdgcn_wmma_f32_16x16x32_bf16(
            false, af1[t], false, bfrag, (short)0, acc[t], false, false);
    }
  }

  // ---- epilogue: C lane col = pixel m (coalesced along w), row = o ----
  int m  = mBlock + mg * 16 + l16;
  int b  = m / HW_;
  int hw = m - b * HW_;
  float* obase = out + b * (O_ * HW_) + hw;
#pragma unroll
  for (int t = 0; t < 8; ++t) {
#pragma unroll
    for (int j = 0; j < 8; ++j) {
      int o = oBase + t * 16 + hl * 8 + j; // C: VGPR j -> M = j + 8*half
      obase[o * HW_] = acc[t][j] + 2.0f * bias[o];  // bias added twice (ref)
    }
  }
}

// ---------------------------------------------------------------------------
extern "C" void kernel_launch(void* const* d_in, const int* in_sizes, int n_in,
                              void* d_out, int out_size, void* d_ws, size_t ws_size,
                              hipStream_t stream) {
  const float* x    = (const float*)d_in[0];
  const float* S1s  = (const float*)d_in[1];
  const float* U1s  = (const float*)d_in[2];
  const float* U2s  = (const float*)d_in[3];
  const float* S2s  = (const float*)d_in[4];
  const float* bias = (const float*)d_in[5];
  float* out = (float*)d_out;

  // workspace: Wt bf16 [O][D] (~0.59 MB) | xpad bf16 NHWC [B][58][58][C] (~27.6 MB)
  __bf16* Wt   = (__bf16*)d_ws;
  size_t wtBytes = (size_t)O_ * D_ * sizeof(__bf16);   // 589824, 256-aligned
  __bf16* xpad = (__bf16*)((char*)d_ws + wtBytes);

  {
    int n = O_ * D_;
    build_weight<<<(n + 255) / 256, 256, 0, stream>>>(S1s, U1s, U2s, S2s, Wt);
  }
  pad_nhwc<<<B_ * HP_, 256, 0, stream>>>(x, xpad);
  skconv_gemm<<<NPIX_ / MBLK, 256, 0, stream>>>(Wt, xpad, bias, out);
}
